// EvidentialUncertaintyDistance_53824530153581
// MI455X (gfx1250) — compile-verified, matrix operand order checked by reading, not run on hardware
//
#include <hip/hip_runtime.h>
#include <hip/hip_bf16.h>

typedef __attribute__((ext_vector_type(16))) _Float16 v16h;
typedef __attribute__((ext_vector_type(8)))  _Float16 v8h;
typedef __attribute__((ext_vector_type(8)))  float    v8f;
typedef __attribute__((ext_vector_type(4)))  float    v4f;

constexpr int kD  = 128;
constexpr int kNQ = 2048;
constexpr int kNP = 256;
constexpr int kH1 = 128;
constexpr int kH2 = 64;

// ---------------------------------------------------------------------------
// out[r*128 + o] = dot(X[r, 0:128], W[o, colOff : colOff+128]) (+ bias[o])
// stored as f16.  W is torch-Linear layout [128, 256].
// ---------------------------------------------------------------------------
__global__ void proj_h16_kernel(const float* __restrict__ X,
                                const float* __restrict__ W,
                                const float* __restrict__ bias,
                                _Float16* __restrict__ out,
                                int nRows, int colOff) {
  int t = blockIdx.x * blockDim.x + threadIdx.x;
  if (t >= nRows * kH1) return;
  int r = t >> 7;
  int o = t & 127;
  const float* x = X + r * kD;
  const float* w = W + o * (2 * kD) + colOff;
  float acc = 0.f;
#pragma unroll
  for (int k = 0; k < kD; k += 4) {
    v4f xv = *(const v4f*)(x + k);
    v4f wv = *(const v4f*)(w + k);
    acc = fmaf(xv[0], wv[0], acc);
    acc = fmaf(xv[1], wv[1], acc);
    acc = fmaf(xv[2], wv[2], acc);
    acc = fmaf(xv[3], wv[3], acc);
  }
  if (bias) acc += bias[o];
  out[t] = (_Float16)acc;
}

// ---------------------------------------------------------------------------
// base[q*256 + p] = sum_k (q_k - p_k)^2   (== max(qn+pn-2qp, 0) exactly)
// ---------------------------------------------------------------------------
__global__ void dist2_kernel(const float* __restrict__ Q,
                             const float* __restrict__ P,
                             float* __restrict__ base) {
  int t = blockIdx.x * blockDim.x + threadIdx.x;
  if (t >= kNQ * kNP) return;
  int qi = t >> 8;
  int pi = t & 255;
  const float* q = Q + qi * kD;
  const float* p = P + pi * kD;
  float acc = 0.f;
#pragma unroll
  for (int k = 0; k < kD; k += 4) {
    v4f qv = *(const v4f*)(q + k);
    v4f pv = *(const v4f*)(p + k);
    float d0 = qv[0] - pv[0];
    float d1 = qv[1] - pv[1];
    float d2 = qv[2] - pv[2];
    float d3 = qv[3] - pv[3];
    acc = fmaf(d0, d0, acc);
    acc = fmaf(d1, d1, acc);
    acc = fmaf(d2, d2, acc);
    acc = fmaf(d3, d3, acc);
  }
  base[t] = acc;
}

// ---------------------------------------------------------------------------
// Main fused kernel.  One wave per query row; 16 prototype tiles of 16.
// GEMM: A[m,k] = relu(qh[qi,k] + phb[p0+m,k])  (16x32 f16 fragments),
//       B[k,n] = W2[n, k]                       (32x16 f16 fragments),
//       D = A x B + C accumulated f32 via v_wmma_f32_16x16x32_f16.
// phb staged in LDS with padded rows (136 f16 = 272B) to spread banks.
// ---------------------------------------------------------------------------
__global__ __launch_bounds__(256) void evid_main_kernel(
    const _Float16* __restrict__ qh16,   // [2048][128]
    const _Float16* __restrict__ phb16,  // [256][128]  (b1 folded in)
    const float* __restrict__ base,      // [2048][256]
    const float* __restrict__ W2,        // [64][128] f32
    const float* __restrict__ b2,        // [64]
    const float* __restrict__ W3,        // [64]  (W3[0])
    const float* __restrict__ b3p,       // [1]
    float* __restrict__ out) {           // [2048][256]
  constexpr int ROWP = 136;  // padded row stride in f16 elements
  __shared__ _Float16 phbL[kNP * ROWP];  // 69,632 B of the 320 KB LDS

  int tid = threadIdx.x;
  // Stage phb16 -> LDS (16B chunks), 4096 chunks across 256 threads.
  for (int c = tid; c < kNP * 16; c += 256) {
    int row = c >> 4;
    int ci = c & 15;
    *(v8h*)&phbL[row * ROWP + ci * 8] = *(const v8h*)&phb16[row * kD + ci * 8];
  }
  __syncthreads();

  int lane = tid & 31;
  int wid  = tid >> 5;
  int n = lane & 15;   // column / row-in-tile index
  int h = lane >> 4;   // lane-half selects K sub-range per ISA layout
  int qi = blockIdx.x * 8 + wid;

  // --- W2 B-fragments, pre-swizzled into registers (one-time). ---
  // b[i] (i<8):  k = kt*32 + h*8 + i ; b[i] (i>=8): k = kt*32 + 16 + h*8 + (i-8)
  v16h w2f[4][4];
#pragma unroll
  for (int nt = 0; nt < 4; ++nt) {
#pragma unroll
    for (int kt = 0; kt < 4; ++kt) {
      const float* wr = W2 + (nt * 16 + n) * kH1 + kt * 32 + h * 8;
      v4f f0 = *(const v4f*)(wr + 0);
      v4f f1 = *(const v4f*)(wr + 4);
      v4f f2 = *(const v4f*)(wr + 16);
      v4f f3 = *(const v4f*)(wr + 20);
      v16h b;
#pragma unroll
      for (int i = 0; i < 4; ++i) b[i]      = (_Float16)f0[i];
#pragma unroll
      for (int i = 0; i < 4; ++i) b[4 + i]  = (_Float16)f1[i];
#pragma unroll
      for (int i = 0; i < 4; ++i) b[8 + i]  = (_Float16)f2[i];
#pragma unroll
      for (int i = 0; i < 4; ++i) b[12 + i] = (_Float16)f3[i];
      w2f[nt][kt] = b;
    }
  }

  // --- qh A-side broadcast fragments (same for every m in the tile). ---
  v16h qf[4];
#pragma unroll
  for (int kt = 0; kt < 4; ++kt) {
    v8h lo = *(const v8h*)(qh16 + qi * kD + kt * 32 + h * 8);
    v8h hi = *(const v8h*)(qh16 + qi * kD + kt * 32 + 16 + h * 8);
    qf[kt] = __builtin_shufflevector(lo, hi, 0, 1, 2, 3, 4, 5, 6, 7,
                                             8, 9, 10, 11, 12, 13, 14, 15);
  }

  float b2l[4], w3l[4];
#pragma unroll
  for (int nt = 0; nt < 4; ++nt) {
    b2l[nt] = b2[nt * 16 + n];
    w3l[nt] = W3[nt * 16 + n];
  }
  float b3v = b3p[0];
  const v16h zero16 = {};

  for (int p0 = 0; p0 < kNP; p0 += 16) {
    v8f acc[4] = {v8f{}, v8f{}, v8f{}, v8f{}};
#pragma unroll
    for (int kt = 0; kt < 4; ++kt) {
      // Build A fragment: a[m, k] = relu(phb[p0+m, k] + qh[qi, k])
      int off = (p0 + n) * ROWP + kt * 32 + h * 8;
      v8h lo = *(const v8h*)&phbL[off];
      v8h hi = *(const v8h*)&phbL[off + 16];
      v16h a = __builtin_shufflevector(lo, hi, 0, 1, 2, 3, 4, 5, 6, 7,
                                               8, 9, 10, 11, 12, 13, 14, 15);
      a = a + qf[kt];
      a = __builtin_elementwise_max(a, zero16);
#pragma unroll
      for (int nt = 0; nt < 4; ++nt) {
        acc[nt] = __builtin_amdgcn_wmma_f32_16x16x32_f16(
            false, a, false, w2f[nt][kt], (short)0, acc[nt], false, false);
      }
    }

    // Epilogue: h2 = relu(acc + b2); s = h2 . W3 reduced across lanes.
    // acc[nt] element r belongs to pair m = r + 8*h, output o = nt*16 + n.
    float s[8];
#pragma unroll
    for (int r = 0; r < 8; ++r) s[r] = 0.f;
#pragma unroll
    for (int nt = 0; nt < 4; ++nt) {
#pragma unroll
      for (int r = 0; r < 8; ++r) {
        float v = fmaxf(acc[nt][r] + b2l[nt], 0.f);
        s[r] = fmaf(v, w3l[nt], s[r]);
      }
    }
    // Reduce over the 16 lanes of each half (xor 1,2,4,8 stays in-half).
#pragma unroll
    for (int off = 8; off >= 1; off >>= 1) {
#pragma unroll
      for (int r = 0; r < 8; ++r) s[r] += __shfl_xor(s[r], off, 32);
    }

    if (n == 0) {  // lanes 0 (pairs p0..p0+7) and 16 (pairs p0+8..p0+15)
      int idx = qi * kNP + p0 + h * 8;
      v4f bl0 = *(const v4f*)(base + idx);
      v4f bl1 = *(const v4f*)(base + idx + 4);
      float ov[8];
#pragma unroll
      for (int r = 0; r < 8; ++r) {
        float e = s[r] + b3v;
        // stable softplus
        float sp = fmaxf(e, 0.f) + log1pf(expf(-fabsf(e)));
        float alpha = sp + 1.0f;
        float conf = 1.0f / (1.0f / alpha + 1e-8f);
        float b = (r < 4) ? bl0[r & 3] : bl1[r & 3];
        ov[r] = b * conf;
      }
      v4f o0 = {ov[0], ov[1], ov[2], ov[3]};
      v4f o1 = {ov[4], ov[5], ov[6], ov[7]};
      *(v4f*)(out + idx) = o0;
      *(v4f*)(out + idx + 4) = o1;
    }
  }
}

// ---------------------------------------------------------------------------
extern "C" void kernel_launch(void* const* d_in, const int* in_sizes, int n_in,
                              void* d_out, int out_size, void* d_ws, size_t ws_size,
                              hipStream_t stream) {
  const float* q  = (const float*)d_in[0];
  const float* p  = (const float*)d_in[1];
  const float* W1 = (const float*)d_in[2];
  const float* b1 = (const float*)d_in[3];
  const float* W2 = (const float*)d_in[4];
  const float* b2 = (const float*)d_in[5];
  const float* W3 = (const float*)d_in[6];
  const float* b3 = (const float*)d_in[7];
  float* out = (float*)d_out;

  char* ws = (char*)d_ws;
  _Float16* qh16  = (_Float16*)(ws);                 // 2048*128*2 = 524288 B
  _Float16* phb16 = (_Float16*)(ws + 524288);        //  256*128*2 =  65536 B
  float*    base  = (float*)   (ws + 589824);        // 2048*256*4 = 2097152 B

  proj_h16_kernel<<<(kNQ * kH1 + 255) / 256, 256, 0, stream>>>(
      q, W1, nullptr, qh16, kNQ, 0);
  proj_h16_kernel<<<(kNP * kH1 + 255) / 256, 256, 0, stream>>>(
      p, W1, b1, phb16, kNP, kD);
  dist2_kernel<<<(kNQ * kNP + 255) / 256, 256, 0, stream>>>(q, p, base);
  evid_main_kernel<<<kNQ / 8, 256, 0, stream>>>(
      qh16, phb16, base, W2, b2, W3, b3, out);
}
